// Decoder_53506702573920
// MI455X (gfx1250) — matrix operand order repaired and checked by reading, not compiled, not run on hardware
//
#include <hip/hip_runtime.h>
#include <hip/hip_bf16.h>

typedef __attribute__((ext_vector_type(16))) _Float16     v16h;
typedef __attribute__((ext_vector_type(8)))  float        v8f;
typedef __attribute__((ext_vector_type(4)))  unsigned int u32x4;
typedef __attribute__((ext_vector_type(4)))  _Float16     half4v;

union FragAB { v16h v; u32x4 q[2]; };

#define T_STEPS 2048

__device__ __forceinline__ float sigmoidf_(float x) { return 1.f / (1.f + __expf(-x)); }

__device__ __forceinline__ float wave_reduce(float v) {
#pragma unroll
    for (int off = 16; off > 0; off >>= 1) v += __shfl_xor(v, off, 32);
    return v;
}

// Generation-counter grid barrier (agent scope, capture-safe, s_sleep spin).
__device__ __forceinline__ void grid_barrier(unsigned* bar, unsigned* gen, unsigned nblocks) {
    __syncthreads();
    if (threadIdx.x == 0) {
        __threadfence();
        unsigned g = __hip_atomic_load(gen, __ATOMIC_RELAXED, __HIP_MEMORY_SCOPE_AGENT);
        unsigned prev = __hip_atomic_fetch_add(bar, 1u, __ATOMIC_ACQ_REL, __HIP_MEMORY_SCOPE_AGENT);
        if (prev == nblocks - 1u) {
            __hip_atomic_store(bar, 0u, __ATOMIC_RELAXED, __HIP_MEMORY_SCOPE_AGENT);
            __hip_atomic_fetch_add(gen, 1u, __ATOMIC_RELEASE, __HIP_MEMORY_SCOPE_AGENT);
        } else {
            while (__hip_atomic_load(gen, __ATOMIC_ACQUIRE, __HIP_MEMORY_SCOPE_AGENT) == g)
                __builtin_amdgcn_s_sleep(2);
        }
    }
    __syncthreads();
}

// ---------------------------------------------------------------- init / convert
__global__ void init_state_kernel(unsigned* bar, float* h1buf, float* h2buf) {
    int t = threadIdx.x;
    if (t < 8) bar[t] = 0u;
    for (int i = t; i < 2 * 1024; i += 256) h1buf[i] = 0.f;
    for (int i = t; i < 2 * 2048; i += 256) h2buf[i] = 0.f;
}

__global__ void f32_to_f16_kernel(const float* __restrict__ src, _Float16* __restrict__ dst, int n) {
    int i = blockIdx.x * blockDim.x + threadIdx.x;
    int stride = gridDim.x * blockDim.x;
    for (; i < n; i += stride) dst[i] = (_Float16)src[i];
}

// ---------------------------------------------------------------- gx1 = Wih1 @ x + b1   (4096x1024 GEMV, once)
__global__ __launch_bounds__(256) void gemv_gx1_kernel(const float* __restrict__ Wih1,
                                                       const float* __restrict__ x,
                                                       const float* __restrict__ b1,
                                                       float* __restrict__ gx1) {
    const int lane = threadIdx.x & 31;
    const int row  = (blockIdx.x << 3) | (threadIdx.x >> 5);  // 512 blocks * 8 waves = 4096 rows
    const float* w = Wih1 + (size_t)row * 1024;
    float acc = 0.f;
    for (int k = lane; k < 1024; k += 32) acc += w[k] * x[k];
    acc = wave_reduce(acc);
    if (lane == 0) gx1[row] = acc + b1[row];
}

// ---------------------------------------------------------------- layer-1 persistent recurrence
// 128 blocks * 8 waves = 1024 waves; wave j owns hidden unit j (c_j in a register).
__global__ __launch_bounds__(256) void lstm1_kernel(const _Float16* __restrict__ Whh,  // [4096,1024] f16
                                                    const float* __restrict__ gx,      // [4096]
                                                    float* __restrict__ hbuf,          // [2][1024] ping-pong
                                                    _Float16* __restrict__ H1,         // [2048,1024] f16
                                                    unsigned* bar, unsigned* gen) {
    __shared__ float hsh[1024];
    const int lane = threadIdx.x & 31;
    const int unit = (blockIdx.x << 3) | (threadIdx.x >> 5);
    const _Float16* w0 = Whh + (size_t)(unit) * 1024;
    const _Float16* w1 = Whh + (size_t)(1024 + unit) * 1024;
    const _Float16* w2 = Whh + (size_t)(2048 + unit) * 1024;
    const _Float16* w3 = Whh + (size_t)(3072 + unit) * 1024;
    const float bi = gx[unit], bff = gx[1024 + unit], bg = gx[2048 + unit], bo = gx[3072 + unit];
    float c = 0.f;
#pragma unroll 1
    for (int t = 0; t < T_STEPS; ++t) {
        const float* hin = hbuf + ((t & 1) << 10);
        for (int i = threadIdx.x; i < 1024; i += 256) hsh[i] = hin[i];
        __syncthreads();
        float a0 = 0.f, a1 = 0.f, a2 = 0.f, a3 = 0.f;
#pragma unroll 4
        for (int k = lane << 2; k < 1024; k += 128) {
            __builtin_prefetch((const void*)(w0 + k + 128), 0, 0);
            const float e0 = hsh[k], e1 = hsh[k + 1], e2 = hsh[k + 2], e3 = hsh[k + 3];
            half4v q0 = *(const half4v*)(w0 + k);
            half4v q1 = *(const half4v*)(w1 + k);
            half4v q2 = *(const half4v*)(w2 + k);
            half4v q3 = *(const half4v*)(w3 + k);
            a0 += (float)q0[0] * e0 + (float)q0[1] * e1 + (float)q0[2] * e2 + (float)q0[3] * e3;
            a1 += (float)q1[0] * e0 + (float)q1[1] * e1 + (float)q1[2] * e2 + (float)q1[3] * e3;
            a2 += (float)q2[0] * e0 + (float)q2[1] * e1 + (float)q2[2] * e2 + (float)q2[3] * e3;
            a3 += (float)q3[0] * e0 + (float)q3[1] * e1 + (float)q3[2] * e2 + (float)q3[3] * e3;
        }
        a0 = wave_reduce(a0); a1 = wave_reduce(a1); a2 = wave_reduce(a2); a3 = wave_reduce(a3);
        if (lane == 0) {
            float gi = sigmoidf_(bi + a0);
            float gf = sigmoidf_(bff + a1);
            float gg = tanhf(bg + a2);
            float go = sigmoidf_(bo + a3);
            c = gf * c + gi * gg;
            float h = go * tanhf(c);
            hbuf[(((t + 1) & 1) << 10) + unit] = h;
            H1[((size_t)t << 10) + unit] = (_Float16)h;
        }
        grid_barrier(bar, gen, gridDim.x);
    }
}

// ---------------------------------------------------------------- GX2 = H1 @ Wih2^T  (2048x8192x1024 WMMA f16 GEMM)
// Wave tile: 16(M) x 64(N); software-pipelined k-loop with register double-buffering
// so each WMMA stage overlaps the next stage's 10 global loads (counted waits, not full drains).

__device__ __forceinline__ void load_a_frag(FragAB& a, const _Float16* aBase, int k0) {
    a.q[0] = *(const u32x4*)(aBase + k0);
    a.q[1] = *(const u32x4*)(aBase + k0 + 16);
}

__device__ __forceinline__ void load_b_frags(FragAB* b, const _Float16* bBase, int k0) {
#pragma unroll
    for (int tt = 0; tt < 4; ++tt) {
        const _Float16* bp = bBase + (size_t)tt * (16 * 1024) + k0;
        b[tt].q[0] = *(const u32x4*)(bp);
        b[tt].q[1] = *(const u32x4*)(bp + 16);
    }
}

__global__ __launch_bounds__(256) void gemm_gx2_kernel(const _Float16* __restrict__ A,  // H1 [2048,1024]
                                                       const _Float16* __restrict__ B,  // Wih2 [8192,1024]
                                                       float* __restrict__ D) {         // [2048,8192]
    const int lane  = threadIdx.x & 31;
    const int wid   = (blockIdx.x << 3) | (threadIdx.x >> 5);  // 0..16383
    const int mIdx  = wid >> 7;   // 0..127 -> 16 rows
    const int nIdx  = wid & 127;  // 0..127 -> 64 cols
    const int m0    = mIdx << 4;
    const int n0    = nIdx << 6;
    const int lhalf = lane >> 4;
    const int lmod  = lane & 15;

    v8f acc[4] = {};
    // A layout (16-bit 16x32): lanes 0-15 hold K {0..7,16..23}, lanes 16-31 hold K {8..15,24..31}.
    const _Float16* aBase = A + (size_t)(m0 + lmod) * 1024 + (lhalf << 3);
    // B layout (16-bit 32x16): lane = N (mod 16); lanes 0-15 hold K {0..7,16..23}, lanes 16-31 hold K {8..15,24..31}.
    const _Float16* bBase = B + (size_t)(n0 + lmod) * 1024 + (lhalf << 3);

    FragAB aP, aQ;
    FragAB bP[4], bQ[4];
    load_a_frag(aP, aBase, 0);
    load_b_frags(bP, bBase, 0);

#pragma unroll 1
    for (int k0 = 0; k0 < 1024; k0 += 64) {
        // stage Q loads (k0+32) in flight while stage P computes
        load_a_frag(aQ, aBase, k0 + 32);
        load_b_frags(bQ, bBase, k0 + 32);
#pragma unroll
        for (int tt = 0; tt < 4; ++tt)
            acc[tt] = __builtin_amdgcn_wmma_f32_16x16x32_f16(false, aP.v, false, bP[tt].v,
                                                             (short)0, acc[tt], false, false);
        // stage P loads (k0+64; harmless redundant reload of k=0 on the last trip)
        const int k2 = (k0 + 64 < 1024) ? (k0 + 64) : 0;
        load_a_frag(aP, aBase, k2);
        load_b_frags(bP, bBase, k2);
#pragma unroll
        for (int tt = 0; tt < 4; ++tt)
            acc[tt] = __builtin_amdgcn_wmma_f32_16x16x32_f16(false, aQ.v, false, bQ[tt].v,
                                                             (short)0, acc[tt], false, false);
    }

    // D layout: VGPR r -> M = r (lanes 0-15) / r+8 (lanes 16-31); N = lane & 15.
    const int mbase = m0 + (lhalf << 3);
#pragma unroll
    for (int tt = 0; tt < 4; ++tt) {
        const int n = n0 + (tt << 4) + lmod;
#pragma unroll
        for (int r = 0; r < 8; ++r) D[(size_t)(mbase + r) * 8192 + n] = acc[tt][r];
    }
}

// ---------------------------------------------------------------- layer-2 persistent recurrence
// 256 blocks * 8 waves = 2048 waves; wave j owns hidden unit j.
__global__ __launch_bounds__(256) void lstm2_kernel(const _Float16* __restrict__ Whh,  // [8192,2048] f16
                                                    const float* __restrict__ GX2,     // [2048,8192]
                                                    const float* __restrict__ b2,      // [8192]
                                                    float* __restrict__ hbuf,          // [2][2048]
                                                    float* __restrict__ H2,            // [2048,2048]
                                                    unsigned* bar, unsigned* gen) {
    __shared__ float hsh[2048];
    const int lane = threadIdx.x & 31;
    const int unit = (blockIdx.x << 3) | (threadIdx.x >> 5);
    const _Float16* w0 = Whh + (size_t)(unit) * 2048;
    const _Float16* w1 = Whh + (size_t)(2048 + unit) * 2048;
    const _Float16* w2 = Whh + (size_t)(4096 + unit) * 2048;
    const _Float16* w3 = Whh + (size_t)(6144 + unit) * 2048;
    const float bi = b2[unit], bff = b2[2048 + unit], bg = b2[4096 + unit], bo = b2[6144 + unit];
    float c = 0.f;
#pragma unroll 1
    for (int t = 0; t < T_STEPS; ++t) {
        const float* hin = hbuf + ((t & 1) << 11);
        for (int i = threadIdx.x; i < 2048; i += 256) hsh[i] = hin[i];
        __syncthreads();
        float a0 = 0.f, a1 = 0.f, a2 = 0.f, a3 = 0.f;
#pragma unroll 4
        for (int k = lane << 2; k < 2048; k += 128) {
            __builtin_prefetch((const void*)(w0 + k + 128), 0, 0);
            const float e0 = hsh[k], e1 = hsh[k + 1], e2 = hsh[k + 2], e3 = hsh[k + 3];
            half4v q0 = *(const half4v*)(w0 + k);
            half4v q1 = *(const half4v*)(w1 + k);
            half4v q2 = *(const half4v*)(w2 + k);
            half4v q3 = *(const half4v*)(w3 + k);
            a0 += (float)q0[0] * e0 + (float)q0[1] * e1 + (float)q0[2] * e2 + (float)q0[3] * e3;
            a1 += (float)q1[0] * e0 + (float)q1[1] * e1 + (float)q1[2] * e2 + (float)q1[3] * e3;
            a2 += (float)q2[0] * e0 + (float)q2[1] * e1 + (float)q2[2] * e2 + (float)q2[3] * e3;
            a3 += (float)q3[0] * e0 + (float)q3[1] * e1 + (float)q3[2] * e2 + (float)q3[3] * e3;
        }
        a0 = wave_reduce(a0); a1 = wave_reduce(a1); a2 = wave_reduce(a2); a3 = wave_reduce(a3);
        if (lane == 0) {
            const float* gxr = GX2 + ((size_t)t << 13);
            float gi = sigmoidf_(gxr[unit] + bi + a0);
            float gf = sigmoidf_(gxr[2048 + unit] + bff + a1);
            float gg = tanhf(gxr[4096 + unit] + bg + a2);
            float go = sigmoidf_(gxr[6144 + unit] + bo + a3);
            c = gf * c + gi * gg;
            float h = go * tanhf(c);
            hbuf[(((t + 1) & 1) << 11) + unit] = h;
            H2[((size_t)t << 11) + unit] = h;
        }
        grid_barrier(bar, gen, gridDim.x);
    }
}

// ---------------------------------------------------------------- out = H2 @ Wout^T + bout
__global__ __launch_bounds__(256) void out_kernel(const float* __restrict__ H2,
                                                  const float* __restrict__ Wout,
                                                  const float* __restrict__ bout,
                                                  float* __restrict__ out) {
    const int lane = threadIdx.x & 31;
    const int t    = (blockIdx.x << 3) | (threadIdx.x >> 5);  // 2048
    const float* h = H2 + ((size_t)t << 11);
    float acc = 0.f;
    for (int j = lane; j < 2048; j += 32) acc += h[j] * Wout[j];
    acc = wave_reduce(acc);
    if (lane == 0) out[t] = acc + bout[0];
}

// ---------------------------------------------------------------- launch
extern "C" void kernel_launch(void* const* d_in, const int* in_sizes, int n_in,
                              void* d_out, int out_size, void* d_ws, size_t ws_size,
                              hipStream_t stream) {
    const float* x    = (const float*)d_in[0];
    const float* Wih1 = (const float*)d_in[1];
    const float* Whh1 = (const float*)d_in[2];
    const float* b1   = (const float*)d_in[3];
    const float* Wih2 = (const float*)d_in[4];
    const float* Whh2 = (const float*)d_in[5];
    const float* b2   = (const float*)d_in[6];
    const float* Wout = (const float*)d_in[7];
    const float* bout = (const float*)d_in[8];
    float* out = (float*)d_out;
    (void)in_sizes; (void)n_in; (void)out_size; (void)ws_size;

    char* ws = (char*)d_ws;
    size_t off = 0;
    auto wsalloc = [&](size_t bytes) -> void* {
        off = (off + 255) & ~(size_t)255;
        void* p = ws + off;
        off += bytes;
        return p;
    };
    unsigned* bar   = (unsigned*)wsalloc(256);                         // [0]=count, [1]=generation
    float*    gx1   = (float*)wsalloc((size_t)4096 * 4);
    float*    h1buf = (float*)wsalloc((size_t)2 * 1024 * 4);
    float*    h2buf = (float*)wsalloc((size_t)2 * 2048 * 4);
    _Float16* H1    = (_Float16*)wsalloc((size_t)2048 * 1024 * 2);     // 4 MB
    float*    GX2   = (float*)wsalloc((size_t)2048 * 8192 * 4);        // 64 MB
    float*    H2    = (float*)wsalloc((size_t)2048 * 2048 * 4);        // 16 MB
    _Float16* Whh1h = (_Float16*)wsalloc((size_t)4096 * 1024 * 2);     // 8 MB
    _Float16* Wih2h = (_Float16*)wsalloc((size_t)8192 * 1024 * 2);     // 16 MB
    _Float16* Whh2h = (_Float16*)wsalloc((size_t)8192 * 2048 * 2);     // 32 MB

    init_state_kernel<<<1, 256, 0, stream>>>(bar, h1buf, h2buf);
    f32_to_f16_kernel<<<1024, 256, 0, stream>>>(Whh1, Whh1h, 4096 * 1024);
    f32_to_f16_kernel<<<1024, 256, 0, stream>>>(Wih2, Wih2h, 8192 * 1024);
    f32_to_f16_kernel<<<2048, 256, 0, stream>>>(Whh2, Whh2h, 8192 * 2048);
    gemv_gx1_kernel<<<512, 256, 0, stream>>>(Wih1, x, b1, gx1);
    lstm1_kernel<<<128, 256, 0, stream>>>(Whh1h, gx1, h1buf, H1, bar, bar + 1);
    gemm_gx2_kernel<<<2048, 256, 0, stream>>>(H1, Wih2h, GX2);
    lstm2_kernel<<<256, 256, 0, stream>>>(Whh2h, GX2, b2, h2buf, H2, bar, bar + 1);
    out_kernel<<<256, 256, 0, stream>>>(H2, Wout, bout, out);
}
